// FlowRefineNetA_20237885899117
// MI455X (gfx1250) — matrix operand order repaired and checked by reading, not compiled, not run on hardware
//
#include <hip/hip_runtime.h>
#include <cstdint>
#include <cstddef>

// ---------------------------------------------------------------------------
// FlowRefineNet for MI455X (gfx1250, wave32).
//  - All conv layers run as implicit-GEMM on v_wmma_f32_16x16x32_bf16.
//    Conv kernel is templated on <TAPS, CPT(=Cpad/32), NM(oc tiles)> so the
//    K-loop fully unrolls (no runtime division) and one wave computes NM
//    16x16 tiles, reusing each activation B-fragment for NM WMMAs.
//  - Activations channels-last, Cpad % 32 == 0: B-fragment = 2 contiguous
//    b128 loads per lane. Weights pre-packed once per launch into the ISA
//    16x32 bf16 A-fragment lane layout (CDNA5 ISA 7.12.2).
//  - Warp / occlusion-MLP / correlation are gather+VALU kernels; x0/x1
//    (75 MB) and all activations fit in the 192 MB L2.
// ---------------------------------------------------------------------------

#define BN    4
#define CCTX  64
#define HH    192
#define WW    192
#define HWP   (HH * WW)
#define NPIX  (BN * HWP)
#define NITER 4

typedef __attribute__((ext_vector_type(16))) __bf16 v16bf;
typedef __attribute__((ext_vector_type(8)))  __bf16 v8bf;
typedef __attribute__((ext_vector_type(8)))  float  v8f;

// ---------------------------------------------------------------------------
// Weight pre-pack: OIHW fp32 -> bf16 A-fragments.
// Fragment layout: [mtile][ktile][lane][e], e=0..15, matching the
// 16-bit 16x32 A-matrix VGPR layout (ISA 7.12.2):
//   half = lane>>4, m = lane&15
//   e<8  -> kk = (half? 8:0)  + e
//   e>=8 -> kk = (half?24:16) + (e-8)
// K ordering: k = tap*Cpad + ic  (tap-major so a 32-chunk never straddles taps)
// ---------------------------------------------------------------------------
__global__ __launch_bounds__(256) void k_prepack(
    const float* __restrict__ w, int Cout, int Cin, int taps, int Cpad,
    int nM, __bf16* __restrict__ afrag)
{
  int cpt   = Cpad >> 5;
  int nK    = taps * cpt;
  int total = nM * nK * 512;
  int i = blockIdx.x * blockDim.x + threadIdx.x;
  if (i >= total) return;
  int e     = i & 15;
  int lane  = (i >> 4) & 31;
  int blk   = i >> 9;              // mtile*nK + kt
  int kt    = blk % nK;
  int mtile = blk / nK;
  int half  = lane >> 4;
  int m     = lane & 15;
  int kk    = (e < 8) ? ((half ? 8 : 0) + e) : ((half ? 24 : 16) + (e - 8));
  int k     = kt * 32 + kk;
  int t     = k / Cpad;
  int ic    = k % Cpad;
  int oc    = mtile * 16 + m;
  float val = 0.f;
  if (ic < Cin && oc < Cout) val = w[(oc * Cin + ic) * taps + t];
  afrag[i] = (__bf16)val;
}

// ---------------------------------------------------------------------------
// Implicit-GEMM conv (1x1 or 3x3 pad=1), bf16 WMMA, fully specialized.
// One wave computes NM 16(oc) x 16(pixel) tiles; fully unrolled K-loop;
// wave-uniform interior fast path (no bounds checks).
// act layout: [b][y][x][CIN] bf16 ; output bf16 channels-last slice
// (coOff within CpadOut) and/or fp32 NCHW.
// ---------------------------------------------------------------------------
template <int TAPS, int CPT, int NM>
__global__ __launch_bounds__(256) void k_conv_wmma(
    const __bf16* __restrict__ act,
    const __bf16* __restrict__ afrag,
    const float* __restrict__ bias, const float* __restrict__ alpha,
    int Cout,
    __bf16* __restrict__ outBf, int CpadOut, int coOff,
    float* __restrict__ outF, int relu)
{
  constexpr int CIN = CPT * 32;
  constexpr int NK  = TAPS * CPT;
  constexpr int PT  = NPIX / 16;                 // pixel tiles (W%16==0)
  int wid  = (blockIdx.x * blockDim.x + threadIdx.x) >> 5;
  int lane = threadIdx.x & 31;
  if (wid >= PT) return;                          // wave-uniform exit
  int b    = wid / (HWP / 16);
  int rem  = wid % (HWP / 16);
  int y    = rem / (WW / 16);
  int xt   = rem % (WW / 16);
  int x    = xt * 16 + (lane & 15);
  int half = lane >> 4;

  v16bf zero = {};
  v8f   zacc = {};
  v8f   acc[NM];
#pragma unroll
  for (int m = 0; m < NM; ++m) acc[m] = zacc;

  const __bf16* apl = afrag + (size_t)lane * 16;  // + (m*NK+kt)*512
  __builtin_prefetch(apl, 0, 1);                  // global_prefetch_b8

  bool interior = (TAPS == 1) ||
                  (y >= 1 && y < HH - 1 && xt >= 1 && xt < (WW / 16) - 1);

  if (interior) {
#pragma unroll
    for (int t = 0; t < TAPS; ++t) {
      const int dy = (TAPS == 9) ? (t / 3 - 1) : 0;
      const int dx = (TAPS == 9) ? (t % 3 - 1) : 0;
      const __bf16* rowp =
          act + (size_t)((b * HH + y + dy) * WW + (x + dx)) * CIN + (half << 3);
      __builtin_prefetch(rowp, 0, 1);
#pragma unroll
      for (int cc = 0; cc < CPT; ++cc) {
        const int kt = t * CPT + cc;
        union { v16bf v; v8bf h[2]; } Bf;
        const __bf16* bp = rowp + cc * 32;
        Bf.h[0] = *(const v8bf*)(bp);
        Bf.h[1] = *(const v8bf*)(bp + 16);
#pragma unroll
        for (int m = 0; m < NM; ++m) {
          union { v16bf v; v8bf h[2]; } A;
          const __bf16* ap = apl + (size_t)(m * NK + kt) * 512;
          A.h[0] = *(const v8bf*)(ap);
          A.h[1] = *(const v8bf*)(ap + 8);
          acc[m] = __builtin_amdgcn_wmma_f32_16x16x32_bf16(
              false, A.v, false, Bf.v, (short)0, acc[m], false, false);
        }
      }
    }
  } else {
#pragma unroll
    for (int t = 0; t < TAPS; ++t) {
      const int dy = (TAPS == 9) ? (t / 3 - 1) : 0;
      const int dx = (TAPS == 9) ? (t % 3 - 1) : 0;
      const int yy = y + dy, xx = x + dx;
      const bool ok = (unsigned)yy < (unsigned)HH && (unsigned)xx < (unsigned)WW;
      const __bf16* rowp =
          act + (size_t)((b * HH + yy) * WW + xx) * CIN + (half << 3);
#pragma unroll
      for (int cc = 0; cc < CPT; ++cc) {
        const int kt = t * CPT + cc;
        union { v16bf v; v8bf h[2]; } Bf;
        if (ok) {
          const __bf16* bp = rowp + cc * 32;
          Bf.h[0] = *(const v8bf*)(bp);
          Bf.h[1] = *(const v8bf*)(bp + 16);
        } else {
          Bf.v = zero;                            // zero-padded border
        }
#pragma unroll
        for (int m = 0; m < NM; ++m) {
          union { v16bf v; v8bf h[2]; } A;
          const __bf16* ap = apl + (size_t)(m * NK + kt) * 512;
          A.h[0] = *(const v8bf*)(ap);
          A.h[1] = *(const v8bf*)(ap + 8);
          acc[m] = __builtin_amdgcn_wmma_f32_16x16x32_bf16(
              false, A.v, false, Bf.v, (short)0, acc[m], false, false);
        }
      }
    }
  }

  size_t pix = (size_t)(b * HH + y) * WW + x;
#pragma unroll
  for (int m = 0; m < NM; ++m) {
#pragma unroll
    for (int r = 0; r < 8; ++r) {
      int oc = m * 16 + half * 8 + r;             // C/D layout (ISA 7.12.2)
      if (oc < Cout) {
        float v = acc[m][r] + bias[oc];
        if (relu) { float a = alpha[oc]; v = (v >= 0.f) ? v : a * v; }
        if (outBf) outBf[pix * CpadOut + coOff + oc] = (__bf16)v;
        if (outF)  outF[((size_t)(b * Cout + oc) * HH + y) * WW + x] = v;
      }
    }
  }
}

// ---------------------------------------------------------------------------
// Center-tap bilinear warp of x0/x1 -> fea0/fea1 bf16 slices of the GRU
// inputs; also scatters flow values (bf16) into flow-net inputs + GRU slices.
// ---------------------------------------------------------------------------
__global__ __launch_bounds__(256) void k_warp(
    const float* __restrict__ x0, const float* __restrict__ x1,
    const float* __restrict__ f0, const float* __restrict__ f1,
    __bf16* __restrict__ gru0, __bf16* __restrict__ gru1,
    __bf16* __restrict__ fin0, __bf16* __restrict__ fin1)
{
  int i = blockIdx.x * blockDim.x + threadIdx.x;
  if (i >= NPIX) return;
  int b = i / HWP, pix = i % HWP;
  int y = pix / WW, x = pix % WW;
  float u0 = f0[(b * 2 + 0) * HWP + pix], v0 = f0[(b * 2 + 1) * HWP + pix];
  float u1 = f1[(b * 2 + 0) * HWP + pix], v1 = f1[(b * 2 + 1) * HWP + pix];

  fin0[(size_t)i * 32 + 0] = (__bf16)u0; fin0[(size_t)i * 32 + 1] = (__bf16)v0;
  fin1[(size_t)i * 32 + 0] = (__bf16)u1; fin1[(size_t)i * 32 + 1] = (__bf16)v1;
  gru0[(size_t)i * 192 + 160] = (__bf16)u0; gru0[(size_t)i * 192 + 161] = (__bf16)v0;
  gru1[(size_t)i * 192 + 160] = (__bf16)u1; gru1[(size_t)i * 192 + 161] = (__bf16)v1;

  {
    float fx = fminf(fmaxf((float)x + u0, 0.f), (float)(WW - 1));
    float fy = fminf(fmaxf((float)y + v0, 0.f), (float)(HH - 1));
    int ix = min(max((int)floorf(fx), 0), WW - 2);
    int iy = min(max((int)floorf(fy), 0), HH - 2);
    float wx = fx - ix, wy = fy - iy;
    for (int c = 0; c < CCTX; ++c) {
      const float* pc = x0 + (size_t)(b * CCTX + c) * HWP;
      float a00 = pc[iy * WW + ix],       a01 = pc[iy * WW + ix + 1];
      float a10 = pc[(iy + 1) * WW + ix], a11 = pc[(iy + 1) * WW + ix + 1];
      float r = (1.f - wy) * ((1.f - wx) * a00 + wx * a01)
              + wy * ((1.f - wx) * a10 + wx * a11);
      gru0[(size_t)i * 192 + 64 + c] = (__bf16)r;
    }
  }
  {
    float fx = fminf(fmaxf((float)x + u1, 0.f), (float)(WW - 1));
    float fy = fminf(fmaxf((float)y + v1, 0.f), (float)(HH - 1));
    int ix = min(max((int)floorf(fx), 0), WW - 2);
    int iy = min(max((int)floorf(fy), 0), HH - 2);
    float wx = fx - ix, wy = fy - iy;
    for (int c = 0; c < CCTX; ++c) {
      const float* pc = x1 + (size_t)(b * CCTX + c) * HWP;
      float a00 = pc[iy * WW + ix],       a01 = pc[iy * WW + ix + 1];
      float a10 = pc[(iy + 1) * WW + ix], a11 = pc[(iy + 1) * WW + ix + 1];
      float r = (1.f - wy) * ((1.f - wx) * a00 + wx * a01)
              + wy * ((1.f - wx) * a10 + wx * a11);
      gru1[(size_t)i * 192 + 64 + c] = (__bf16)r;
    }
  }
}

// ---------------------------------------------------------------------------
// Occlusion MLP (1x1 convs 128->32->32->1 + sigmoid) + blended fea.
// One wave32 per pixel: lane j owns hidden unit j; cross-lane via shuffles.
// Writes fea (bf16) into ch 0..63 of both GRU input buffers.
// ---------------------------------------------------------------------------
__global__ __launch_bounds__(256) void k_occl(
    const float* __restrict__ w1, const float* __restrict__ b1, const float* __restrict__ a1,
    const float* __restrict__ w2, const float* __restrict__ b2, const float* __restrict__ a2,
    const float* __restrict__ w3, const float* __restrict__ b3,
    __bf16* __restrict__ gru0, __bf16* __restrict__ gru1)
{
  int wid  = (blockIdx.x * blockDim.x + threadIdx.x) >> 5;
  int lane = threadIdx.x & 31;
  if (wid >= NPIX) return;
  const __bf16* in0 = gru0 + (size_t)wid * 192 + 64;  // fea0
  const __bf16* in1 = gru1 + (size_t)wid * 192 + 64;  // fea1

  float s = b1[lane];
  for (int c = 0; c < 64; ++c) {
    s += w1[lane * 128 + c]      * (float)in0[c];
    s += w1[lane * 128 + 64 + c] * (float)in1[c];
  }
  float h1 = (s >= 0.f) ? s : a1[lane] * s;

  float s2 = b2[lane];
  for (int j = 0; j < 32; ++j)
    s2 += w2[lane * 32 + j] * __shfl(h1, j, 32);
  float h2 = (s2 >= 0.f) ? s2 : a2[lane] * s2;

  float part = w3[lane] * h2;
#pragma unroll
  for (int off = 16; off > 0; off >>= 1) part += __shfl_xor(part, off, 32);
  float occ = 1.f / (1.f + expf(-(part + b3[0])));

  for (int c = lane; c < 64; c += 32) {
    float fe = (float)in0[c] * occ + (float)in1[c] * (1.f - occ);
    gru0[(size_t)wid * 192 + c] = (__bf16)fe;
    gru1[(size_t)wid * 192 + c] = (__bf16)fe;
  }
}

// ---------------------------------------------------------------------------
// 9-point normalized correlation. The 9 bilinear taps share a 4x4 patch per
// channel (rows iy-1..iy+2), with zero padding for the unfolded border.
// ---------------------------------------------------------------------------
__global__ __launch_bounds__(256) void k_corr(
    const float* __restrict__ x0, const float* __restrict__ x1,
    const float* __restrict__ f0, const float* __restrict__ f1,
    const __bf16* __restrict__ gru0,
    __bf16* __restrict__ co0, __bf16* __restrict__ co1)
{
  int i = blockIdx.x * blockDim.x + threadIdx.x;
  if (i >= NPIX) return;
  int b = i / HWP, pix = i % HWP;
  int y = pix / WW, x = pix % WW;

  float u0 = f0[(b * 2 + 0) * HWP + pix], v0 = f0[(b * 2 + 1) * HWP + pix];
  float u1 = f1[(b * 2 + 0) * HWP + pix], v1 = f1[(b * 2 + 1) * HWP + pix];

  float fx0 = fminf(fmaxf((float)x + u0, 0.f), (float)(WW - 1));
  float fy0 = fminf(fmaxf((float)y + v0, 0.f), (float)(HH - 1));
  int ix0 = min(max((int)floorf(fx0), 0), WW - 2);
  int iy0 = min(max((int)floorf(fy0), 0), HH - 2);
  float wx0 = fx0 - ix0, wy0 = fy0 - iy0;

  float fx1 = fminf(fmaxf((float)x + u1, 0.f), (float)(WW - 1));
  float fy1 = fminf(fmaxf((float)y + v1, 0.f), (float)(HH - 1));
  int ix1 = min(max((int)floorf(fx1), 0), WW - 2);
  int iy1 = min(max((int)floorf(fy1), 0), HH - 2);
  float wx1 = fx1 - ix1, wy1 = fy1 - iy1;

  float d0[9] = {}, n0[9] = {}, d1[9] = {}, n1[9] = {}, fn = 0.f;

  for (int c = 0; c < CCTX; ++c) {
    float fe = (float)gru0[(size_t)i * 192 + c];
    fn += fe * fe;
    const float* p0b = x0 + (size_t)(b * CCTX + c) * HWP;
    const float* p1b = x1 + (size_t)(b * CCTX + c) * HWP;
    float p0[16], p1[16];
#pragma unroll
    for (int rr = 0; rr < 4; ++rr)
#pragma unroll
      for (int ss = 0; ss < 4; ++ss) {
        int r0 = iy0 - 1 + rr, c0 = ix0 - 1 + ss;
        p0[rr * 4 + ss] = (r0 >= 0 && r0 < HH && c0 >= 0 && c0 < WW)
                              ? p0b[r0 * WW + c0] : 0.f;
        int r1 = iy1 - 1 + rr, c1 = ix1 - 1 + ss;
        p1[rr * 4 + ss] = (r1 >= 0 && r1 < HH && c1 >= 0 && c1 < WW)
                              ? p1b[r1 * WW + c1] : 0.f;
      }
#pragma unroll
    for (int p = 0; p < 9; ++p) {
      int pi = p / 3, pj = p % 3;
      float q0 = (1.f - wy0) * ((1.f - wx0) * p0[pi * 4 + pj] + wx0 * p0[pi * 4 + pj + 1])
               + wy0 * ((1.f - wx0) * p0[(pi + 1) * 4 + pj] + wx0 * p0[(pi + 1) * 4 + pj + 1]);
      d0[p] += fe * q0;  n0[p] += q0 * q0;
      float q1 = (1.f - wy1) * ((1.f - wx1) * p1[pi * 4 + pj] + wx1 * p1[pi * 4 + pj + 1])
               + wy1 * ((1.f - wx1) * p1[(pi + 1) * 4 + pj] + wx1 * p1[(pi + 1) * 4 + pj + 1]);
      d1[p] += fe * q1;  n1[p] += q1 * q1;
    }
  }
  float rf = rsqrtf(fn + 1e-12f);
#pragma unroll
  for (int p = 0; p < 9; ++p) {
    co0[(size_t)i * 32 + p] = (__bf16)(d0[p] * rf * rsqrtf(n0[p] + 1e-12f));
    co1[(size_t)i * 32 + p] = (__bf16)(d1[p] * rf * rsqrtf(n1[p] + 1e-12f));
  }
}

// ---------------------------------------------------------------------------
__global__ __launch_bounds__(256) void k_copy2(
    const float* __restrict__ a, const float* __restrict__ b,
    float* __restrict__ oa, float* __restrict__ ob, int n)
{
  int i = blockIdx.x * blockDim.x + threadIdx.x;
  if (i < n) { oa[i] = a[i]; ob[i] = b[i]; }
}

__global__ __launch_bounds__(256) void k_update(
    float* __restrict__ flow, const float* __restrict__ delta, int n)
{
  int i = blockIdx.x * blockDim.x + threadIdx.x;
  if (i < n) flow[i] += delta[i];
}

__global__ __launch_bounds__(256) void k_output(
    const float* __restrict__ f0, const float* __restrict__ f1,
    float* __restrict__ out)
{
  int i = blockIdx.x * blockDim.x + threadIdx.x;
  int n = BN * 2 * HWP;
  if (i >= n) return;
  int b = i / (2 * HWP), r = i % (2 * HWP);
  out[(size_t)b * 4 * HWP + r]            = f0[i];
  out[(size_t)b * 4 * HWP + 2 * HWP + r]  = f1[i];
}

// ---------------------------------------------------------------------------
// Host orchestration
// ---------------------------------------------------------------------------
struct LayerCfg { int wi, bi, ai, Cout, Cin, taps, Cpad, nM; };

extern "C" void kernel_launch(void* const* d_in, const int* in_sizes, int n_in,
                              void* d_out, int out_size, void* d_ws, size_t ws_size,
                              hipStream_t stream)
{
  (void)in_sizes; (void)n_in; (void)out_size;
  auto F = [&](int i) { return (const float*)d_in[i]; };

  // input index map (setup_inputs dict order; params: occl,corr,flow,motion,gru,head)
  const float* X0 = F(0);  const float* X1 = F(1);
  const float* FL0 = F(2); const float* FL1 = F(3);
  // occl: w1=4 b1=5 a1=6 w2=7 b2=8 a2=9 w3=10 b3=11
  const LayerCfg L[9] = {
    {12, 13, 14, 32,   9, 1,  32, 2},  // corr conv1
    {15, 16, 17, 32,  32, 1,  32, 2},  // corr conv2
    {18, 19, 20, 32,   2, 9,  32, 2},  // flow conv1
    {21, 22, 23, 32,  32, 9,  32, 2},  // flow conv2
    {24, 25, 26, 32,  64, 9,  64, 2},  // motion
    {27, 28, 29, 32, 162, 9, 192, 2},  // gru conv1
    {30, 31, 32, 32,  32, 9,  32, 2},  // gru conv2
    {33, 34, 35, 32,  32, 9,  32, 2},  // head conv1
    {36, 37, -1,  2,  32, 9,  32, 1},  // head conv2 (fp32 out, no act)
  };

  // ---- workspace carve-up (all 256B aligned) ----
  char* ws = (char*)d_ws;
  size_t off = 0;
  auto carve = [&](size_t bytes) { void* p = ws + off; off = (off + bytes + 255) & ~(size_t)255; return p; };
  __bf16* gru0 = (__bf16*)carve((size_t)NPIX * 192 * 2);
  __bf16* gru1 = (__bf16*)carve((size_t)NPIX * 192 * 2);
  __bf16* cin0 = (__bf16*)carve((size_t)NPIX * 32 * 2);
  __bf16* cin1 = (__bf16*)carve((size_t)NPIX * 32 * 2);
  __bf16* fin0 = (__bf16*)carve((size_t)NPIX * 32 * 2);
  __bf16* fin1 = (__bf16*)carve((size_t)NPIX * 32 * 2);
  __bf16* motin = (__bf16*)carve((size_t)NPIX * 64 * 2);
  __bf16* tmpA = (__bf16*)carve((size_t)NPIX * 32 * 2);
  __bf16* tmpB = (__bf16*)carve((size_t)NPIX * 32 * 2);
  float* flow0c = (float*)carve((size_t)BN * 2 * HWP * 4);
  float* flow1c = (float*)carve((size_t)BN * 2 * HWP * 4);
  float* delta  = (float*)carve((size_t)BN * 2 * HWP * 4);
  __bf16* pk[9];
  int nKL[9];
  for (int l = 0; l < 9; ++l) {
    nKL[l] = L[l].taps * (L[l].Cpad / 32);
    pk[l] = (__bf16*)carve((size_t)L[l].nM * nKL[l] * 512 * 2);
  }
  size_t used = off;
  if (used > ws_size) used = ws_size;  // compile-only guard

  // zero workspace: establishes zero pad channels deterministically each call
  hipMemsetAsync(d_ws, 0, used, stream);

  // fp32 working flows (never mutate d_in)
  {
    int n = BN * 2 * HWP;
    k_copy2<<<(n + 255) / 256, 256, 0, stream>>>(FL0, FL1, flow0c, flow1c, n);
  }

  // pre-pack all conv weights into WMMA A-fragments (once per launch)
  for (int l = 0; l < 9; ++l) {
    int total = L[l].nM * nKL[l] * 512;
    k_prepack<<<(total + 255) / 256, 256, 0, stream>>>(
        F(L[l].wi), L[l].Cout, L[l].Cin, L[l].taps, L[l].Cpad, L[l].nM, pk[l]);
  }

  const int convBlocks = ((NPIX / 16) * 32 + 255) / 256;
  auto conv = [&](int l, const __bf16* act, __bf16* outBf, int CpadOut,
                  int coOff, float* outF, int relu) {
    const float* al = (L[l].ai >= 0) ? F(L[l].ai) : nullptr;
    const float* bi = F(L[l].bi);
    switch (l) {
      case 0: case 1:                   // 1x1, Cpad 32, 2 oc tiles
        k_conv_wmma<1, 1, 2><<<convBlocks, 256, 0, stream>>>(
            act, pk[l], bi, al, L[l].Cout, outBf, CpadOut, coOff, outF, relu);
        break;
      case 2: case 3: case 6: case 7:   // 3x3, Cpad 32, 2 oc tiles
        k_conv_wmma<9, 1, 2><<<convBlocks, 256, 0, stream>>>(
            act, pk[l], bi, al, L[l].Cout, outBf, CpadOut, coOff, outF, relu);
        break;
      case 4:                           // 3x3, Cpad 64, 2 oc tiles
        k_conv_wmma<9, 2, 2><<<convBlocks, 256, 0, stream>>>(
            act, pk[l], bi, al, L[l].Cout, outBf, CpadOut, coOff, outF, relu);
        break;
      case 5:                           // 3x3, Cpad 192, 2 oc tiles
        k_conv_wmma<9, 6, 2><<<convBlocks, 256, 0, stream>>>(
            act, pk[l], bi, al, L[l].Cout, outBf, CpadOut, coOff, outF, relu);
        break;
      default:                          // 3x3, Cpad 32, 1 oc tile (head2)
        k_conv_wmma<9, 1, 1><<<convBlocks, 256, 0, stream>>>(
            act, pk[l], bi, al, L[l].Cout, outBf, CpadOut, coOff, outF, relu);
        break;
    }
  };

  const int pixBlocks  = (NPIX + 255) / 256;
  const int waveBlocks = (NPIX * 32 + 255) / 256;   // wave-per-pixel kernels
  const int nflow      = BN * 2 * HWP;

  for (int it = 0; it < NITER; ++it) {
    // 1) center warps + flow scatter
    k_warp<<<pixBlocks, 256, 0, stream>>>(X0, X1, flow0c, flow1c,
                                          gru0, gru1, fin0, fin1);
    // 2) occlusion MLP + blended fea
    k_occl<<<waveBlocks, 256, 0, stream>>>(F(4), F(5), F(6), F(7), F(8), F(9),
                                           F(10), F(11), gru0, gru1);
    // 3) 9-point normalized correlation
    k_corr<<<pixBlocks, 256, 0, stream>>>(X0, X1, flow0c, flow1c, gru0,
                                          cin0, cin1);
    // 4) per-branch conv stacks (WMMA)
    for (int br = 0; br < 2; ++br) {
      const __bf16* cin = br ? cin1 : cin0;
      const __bf16* fin = br ? fin1 : fin0;
      __bf16* gru = br ? gru1 : gru0;
      float*  flw = br ? flow1c : flow0c;

      conv(0, cin,  tmpA, 32, 0, nullptr, 1);     // corr 9->32
      conv(1, tmpA, motin, 64, 0, nullptr, 1);    // corr 32->32 -> motin[0:32]
      conv(2, fin,  tmpB, 32, 0, nullptr, 1);     // flow 2->32
      conv(3, tmpB, motin, 64, 32, nullptr, 1);   // flow 32->32 -> motin[32:64]
      conv(4, motin, gru, 192, 128, nullptr, 1);  // motion 64->32 -> gru[128:160]
      conv(5, gru,  tmpA, 32, 0, nullptr, 1);     // gru 162(pad192)->32
      conv(6, tmpA, tmpB, 32, 0, nullptr, 1);     // gru 32->32
      conv(7, tmpB, tmpA, 32, 0, nullptr, 1);     // head 32->32
      conv(8, tmpA, nullptr, 0, 0, delta, 0);     // head 32->2, fp32 NCHW

      k_update<<<(nflow + 255) / 256, 256, 0, stream>>>(flw, delta, nflow);
    }
  }

  k_output<<<(nflow + 255) / 256, 256, 0, stream>>>(flow0c, flow1c,
                                                    (float*)d_out);
}
